// SineAttention_28905129902577
// MI455X (gfx1250) — compile-verified, matrix operand order checked by reading, not run on hardware
//
#include <hip/hip_runtime.h>

typedef __bf16 bf16_t;
typedef __attribute__((ext_vector_type(8)))  __bf16 v8bf;
typedef __attribute__((ext_vector_type(16))) __bf16 v16bf;
typedef __attribute__((ext_vector_type(8)))  float  v8f;

#define D_MODEL   1024
#define NUM_HEADS 16
#define DK        64
#define BATCH     2
#define SEQ       2048
#define M_TOTAL   (BATCH * SEQ)   // 4096

// ---------------------------------------------------------------------------
// Fragment loader: CDNA5 16x32 bf16 A/B fragment for one lane.
// Lane L: m/n = L&15, kbase = (L>=16)*8. Elements 0..7 -> K = kbase+0..7,
// elements 8..15 -> K = kbase+16..23. Two aligned 16-byte loads
// (global_load_b128 or ds_load_b128 depending on address space).
// ---------------------------------------------------------------------------
__device__ __forceinline__ v16bf ld_frag(const bf16_t* p) {
    v8bf lo = *(const v8bf*)(p);
    v8bf hi = *(const v8bf*)(p + 16);
    return __builtin_shufflevector(lo, hi, 0, 1, 2, 3, 4, 5, 6, 7,
                                           8, 9, 10, 11, 12, 13, 14, 15);
}

// ---------------------------------------------------------------------------
// CDNA5 async global->LDS copy (16B per lane), tracked by ASYNCcnt.
// the LDS address is taken from the vdst VGPR; low 32 bits of a generic LDS pointer are
// the wave-relative LDS byte offset.
// ---------------------------------------------------------------------------
__device__ __forceinline__ void async_copy16(const bf16_t* g, bf16_t* l) {
    unsigned           lds = (unsigned)(uintptr_t)l;
    unsigned long long ga  = (unsigned long long)(uintptr_t)g;
    asm volatile("global_load_async_to_lds_b128 %0, %1, off"
                 :: "v"(lds), "v"(ga) : "memory");
}
__device__ __forceinline__ void async_wait0() {
    asm volatile("s_wait_asynccnt 0x0" ::: "memory");
}

// ---------------------------------------------------------------------------
// f32 -> bf16 convert
// ---------------------------------------------------------------------------
__global__ void cvt_f32_bf16(const float* __restrict__ src,
                             bf16_t* __restrict__ dst, int n) {
    int i = blockIdx.x * blockDim.x + threadIdx.x;
    if (i < n) dst[i] = (bf16_t)src[i];
}

// ---------------------------------------------------------------------------
// Y = A (MxK, bf16 row-major) * W^T (W is NxK bf16 row-major) + bias(f32)
// MODE 0: store bf16 row-major MxN
// MODE 1: store f32  row-major MxN
// MODE 2: store bf16 per-head transposed: Vt[((b*H+h)*DK+d)*SEQ + s]
// Block: 256 threads (8 waves); wave computes 16 rows x 64 cols.
// W tile (64x32 = 4KB) is staged once per block per k-step via async
// copies to LDS (8x L2 traffic reduction vs per-wave loads).
// Grid: (M/128, N/64).
// ---------------------------------------------------------------------------
template <int MODE>
__global__ __launch_bounds__(256) void gemm_wt(
    const bf16_t* __restrict__ A, const bf16_t* __restrict__ W,
    const float* __restrict__ bias, void* __restrict__ Y,
    int M, int N, int K)
{
    __shared__ bf16_t sW[64][32];   // [n within 64-col tile][k within 32]

    const int tid     = threadIdx.x;
    const int lane    = tid & 31;
    const int wave    = tid >> 5;
    const int rowBase = blockIdx.x * 128 + wave * 16;
    const int colBase = blockIdx.y * 64;

    const int mloc = lane & 15;
    const int kb   = (lane >> 4) * 8;
    const int hi8  = (lane >> 4) * 8;

    // staging assignment: 256 threads x 16B == 4KB tile
    const int cRow = tid >> 2;         // 0..63 (column of Y == row of W)
    const int cChk = (tid & 3) * 8;    // k element offset (8 bf16 = 16B)

    v8f acc[4] = {};

    for (int k0 = 0; k0 < K; k0 += 32) {
        if (k0) __syncthreads();       // previous tile fully consumed
        async_copy16(W + (size_t)(colBase + cRow) * K + k0 + cChk,
                     &sW[cRow][cChk]);
        async_wait0();
        __syncthreads();               // tile visible to all waves

        v16bf af = ld_frag(A + (size_t)(rowBase + mloc) * K + k0 + kb);
#pragma unroll
        for (int t = 0; t < 4; ++t) {
            v16bf bf = ld_frag(&sW[t * 16 + mloc][kb]);
            acc[t] = __builtin_amdgcn_wmma_f32_16x16x32_bf16(
                false, af, false, bf, (short)0, acc[t], false, false);
        }
    }

#pragma unroll
    for (int t = 0; t < 4; ++t) {
#pragma unroll
        for (int i = 0; i < 8; ++i) {
            int row = rowBase + i + hi8;
            int col = colBase + t * 16 + mloc;
            float v = acc[t][i] + bias[col];
            if (MODE == 0) {
                ((bf16_t*)Y)[(size_t)row * N + col] = (bf16_t)v;
            } else if (MODE == 1) {
                ((float*)Y)[(size_t)row * N + col] = v;
            } else {
                int b = row >> 11, s = row & 2047;
                int h = col >> 6,  d = col & 63;
                ((bf16_t*)Y)[((size_t)((b * NUM_HEADS + h) * DK + d)) * SEQ + s] =
                    (bf16_t)v;
            }
        }
    }
}

// ---------------------------------------------------------------------------
// Sine attention (no softmax): O = sin(3.75 * Q K^T) V per (b,h).
// Q,K in (B*S, D_MODEL) bf16 row-major; Vt in ((B*H*DK), SEQ) bf16.
// K chunk (32x64) and V chunk (64x32) staged per block via async copies;
// score tile round-trips wave-private LDS for the C->A fragment transpose.
// Grid: (SEQ/128, B*H); block 256 = 8 waves; wave owns 16 query rows.
// ---------------------------------------------------------------------------
__global__ __launch_bounds__(256) void sine_attn(
    const bf16_t* __restrict__ Q, const bf16_t* __restrict__ Km,
    const bf16_t* __restrict__ Vt, bf16_t* __restrict__ O)
{
    __shared__ bf16_t sK[32][64];        // [key in chunk][d]        4KB
    __shared__ bf16_t sV[64][32];        // [d][key in chunk]        4KB
    __shared__ bf16_t sS[8][16][32];     // per-wave score tile      8KB

    const int tid   = threadIdx.x;
    const int lane  = tid & 31;
    const int wave  = tid >> 5;
    const int bh    = blockIdx.y;            // b*H + h
    const int b     = bh >> 4;
    const int h     = bh & 15;
    const int qBase = blockIdx.x * 128 + wave * 16;

    const int mloc = lane & 15;
    const int kb   = (lane >> 4) * 8;
    const int hi8  = (lane >> 4) * 8;

    // staging assignments (16B per thread per buffer)
    const int kRow = tid >> 3;          // 0..31 key row, 64 elems each
    const int kChk = (tid & 7) * 8;
    const int vRow = tid >> 2;          // 0..63 d row, 32 elems each
    const int vChk = (tid & 3) * 8;

    // Q fragments: 16 rows x 64 (two K=32 steps), kept resident.
    v16bf qf[2];
#pragma unroll
    for (int kk = 0; kk < 2; ++kk)
        qf[kk] = ld_frag(Q + ((size_t)(b * SEQ + qBase + mloc)) * D_MODEL
                           + h * DK + kk * 32 + kb);

    v8f oacc[4] = {};   // 16 x 64 f32 output accumulators

    for (int kc = 0; kc < SEQ; kc += 32) {
        if (kc) __syncthreads();   // previous K/V chunk fully consumed
        async_copy16(Km + ((size_t)(b * SEQ + kc + kRow)) * D_MODEL
                        + h * DK + kChk, &sK[kRow][kChk]);
        async_copy16(Vt + ((size_t)(bh * DK + vRow)) * SEQ + kc + vChk,
                     &sV[vRow][vChk]);
        async_wait0();
        __syncthreads();           // chunk visible to all waves

        // ---- logits for 32 keys: two 16x16 tiles, fp32 accumulate ----
#pragma unroll
        for (int t = 0; t < 2; ++t) {
            v8f lacc = {};
#pragma unroll
            for (int kk = 0; kk < 2; ++kk) {
                v16bf kf = ld_frag(&sK[t * 16 + mloc][kk * 32 + kb]);
                lacc = __builtin_amdgcn_wmma_f32_16x16x32_bf16(
                    false, qf[kk], false, kf, (short)0, lacc, false, false);
            }
            // sin(omega0/sqrt(dk) * logit) = sin(3.75 * logit); stash in LDS
            // to re-layout C-fragment -> A-fragment (cross-lane transpose).
#pragma unroll
            for (int i = 0; i < 8; ++i) {
                float sv = __sinf(3.75f * lacc[i]);
                sS[wave][i + hi8][t * 16 + mloc] = (bf16_t)sv;
            }
        }

        // ---- score fragment (16 q-rows x 32 keys) back from LDS ----
        v16bf sf = ld_frag(&sS[wave][mloc][kb]);

        // ---- accumulate O += S * V ----
#pragma unroll
        for (int t = 0; t < 4; ++t) {
            v16bf vf = ld_frag(&sV[t * 16 + mloc][kb]);
            oacc[t] = __builtin_amdgcn_wmma_f32_16x16x32_bf16(
                false, sf, false, vf, (short)0, oacc[t], false, false);
        }
    }

    // store O rows back in (b, s, h*DK+d) layout (bf16 for final GEMM)
#pragma unroll
    for (int t = 0; t < 4; ++t) {
#pragma unroll
        for (int i = 0; i < 8; ++i) {
            int row = qBase + i + hi8;
            int col = h * DK + t * 16 + mloc;
            O[((size_t)(b * SEQ + row)) * D_MODEL + col] = (bf16_t)oacc[t][i];
        }
    }
}

// ---------------------------------------------------------------------------
// Launch
// ---------------------------------------------------------------------------
extern "C" void kernel_launch(void* const* d_in, const int* in_sizes, int n_in,
                              void* d_out, int out_size, void* d_ws, size_t ws_size,
                              hipStream_t stream) {
    const float* x  = (const float*)d_in[0];
    const float* Wq = (const float*)d_in[1];
    const float* bq = (const float*)d_in[2];
    const float* Wk = (const float*)d_in[3];
    const float* bk = (const float*)d_in[4];
    const float* Wv = (const float*)d_in[5];
    const float* bv = (const float*)d_in[6];
    const float* Wo = (const float*)d_in[7];
    const float* bo = (const float*)d_in[8];

    const size_t nX = (size_t)M_TOTAL * D_MODEL;   // 4.19M elems
    const size_t nW = (size_t)D_MODEL * D_MODEL;   // 1.05M elems

    char* ws = (char*)d_ws;
    size_t off = 0;
    bf16_t* xb  = (bf16_t*)(ws + off); off += nX * sizeof(bf16_t);
    bf16_t* wqb = (bf16_t*)(ws + off); off += nW * sizeof(bf16_t);
    bf16_t* wkb = (bf16_t*)(ws + off); off += nW * sizeof(bf16_t);
    bf16_t* wvb = (bf16_t*)(ws + off); off += nW * sizeof(bf16_t);
    bf16_t* wob = (bf16_t*)(ws + off); off += nW * sizeof(bf16_t);
    bf16_t* Qb  = (bf16_t*)(ws + off); off += nX * sizeof(bf16_t);
    bf16_t* Kb  = (bf16_t*)(ws + off); off += nX * sizeof(bf16_t);
    bf16_t* Vt  = (bf16_t*)(ws + off); off += nX * sizeof(bf16_t);
    bf16_t* Ab  = (bf16_t*)(ws + off); off += nX * sizeof(bf16_t);

    const int cvtB = 256;
    cvt_f32_bf16<<<(int)((nX + cvtB - 1) / cvtB), cvtB, 0, stream>>>(x,  xb,  (int)nX);
    cvt_f32_bf16<<<(int)((nW + cvtB - 1) / cvtB), cvtB, 0, stream>>>(Wq, wqb, (int)nW);
    cvt_f32_bf16<<<(int)((nW + cvtB - 1) / cvtB), cvtB, 0, stream>>>(Wk, wkb, (int)nW);
    cvt_f32_bf16<<<(int)((nW + cvtB - 1) / cvtB), cvtB, 0, stream>>>(Wv, wvb, (int)nW);
    cvt_f32_bf16<<<(int)((nW + cvtB - 1) / cvtB), cvtB, 0, stream>>>(Wo, wob, (int)nW);

    dim3 gGemm(M_TOTAL / 128, D_MODEL / 64);
    gemm_wt<0><<<gGemm, 256, 0, stream>>>(xb, wqb, bq, (void*)Qb,
                                          M_TOTAL, D_MODEL, D_MODEL);
    gemm_wt<0><<<gGemm, 256, 0, stream>>>(xb, wkb, bk, (void*)Kb,
                                          M_TOTAL, D_MODEL, D_MODEL);
    gemm_wt<2><<<gGemm, 256, 0, stream>>>(xb, wvb, bv, (void*)Vt,
                                          M_TOTAL, D_MODEL, D_MODEL);

    dim3 gAttn(SEQ / 128, BATCH * NUM_HEADS);
    sine_attn<<<gAttn, 256, 0, stream>>>(Qb, Kb, Vt, Ab);

    gemm_wt<1><<<gGemm, 256, 0, stream>>>(Ab, wob, bo, d_out,
                                          M_TOTAL, D_MODEL, D_MODEL);
}
